// GptOssExperts_49529562857552
// MI455X (gfx1250) — matrix operand order repaired and checked by reading, not compiled, not run on hardware
//
#include <hip/hip_runtime.h>

#define E_    16
#define T_    8192
#define H_    2880
#define I_    2880
#define KTOP  4
#define TKP   (T_ * KTOP)          // 32768 (token,expert) pairs
#define ALPHA 1.702f
#define LIMIT 7.0f

#define BM    256                   // rows (token slots) per block
#define NTH   512                   // threads per block (16 waves)
#define KT    32                    // K slab per WMMA step
#define CHB   64                    // intermediate channels per block (GEMM1)
#define BN2   64                    // output cols per block (GEMM2)
#define MAXD  160                   // max M-tile descriptors (worst case 143)

#define USE_ASYNC_LDS 1

typedef __attribute__((ext_vector_type(16))) __bf16 v16bf;
typedef __attribute__((ext_vector_type(8)))  float  v8f;

struct __align__(16) Desc { int e, row0, rows, pad; };

__device__ __forceinline__ unsigned short f2bf(float x) {
  unsigned u = __float_as_uint(x);
  u += 0x7fffu + ((u >> 16) & 1u);      // round-to-nearest-even
  return (unsigned short)(u >> 16);
}

// ---------------------------------------------------------------- utilities
__global__ void k_zero_out(float4* __restrict__ out4) {
  size_t i = (size_t)blockIdx.x * blockDim.x + threadIdx.x;
  if (i < (size_t)T_ * H_ / 4) out4[i] = make_float4(0.f, 0.f, 0.f, 0.f);
}

__global__ void k_zero_misc(int* __restrict__ p) {
  if (threadIdx.x < 48) p[threadIdx.x] = 0;   // cnt[16], cursor[16], offs[16]
}

__global__ void k_count(const int* __restrict__ ridx, int* __restrict__ cnt) {
  int g = blockIdx.x * blockDim.x + threadIdx.x;
  if (g < TKP) atomicAdd(&cnt[ridx[g] & 15], 1);
}

__global__ void k_prefix(const int* __restrict__ cnt, int* __restrict__ offs,
                         Desc* __restrict__ desc) {
  if (threadIdx.x != 0 || blockIdx.x != 0) return;
  int off = 0, nd = 0;
  for (int e = 0; e < E_; ++e) {
    offs[e] = off;
    int c = cnt[e];
    for (int bm = 0; bm * BM < c; ++bm) {
      Desc d; d.e = e; d.row0 = off + bm * BM;
      d.rows = (c - bm * BM < BM) ? (c - bm * BM) : BM; d.pad = 0;
      desc[nd++] = d;
    }
    off += c;
  }
  for (; nd < MAXD; ++nd) { Desc d; d.e = -1; d.row0 = 0; d.rows = 0; d.pad = 0; desc[nd] = d; }
}

__global__ void k_scatter(const int* __restrict__ ridx, const int* __restrict__ offs,
                          int* __restrict__ cursor, int* __restrict__ order) {
  int g = blockIdx.x * blockDim.x + threadIdx.x;
  if (g < TKP) {
    int e = ridx[g] & 15;
    int pos = atomicAdd(&cursor[e], 1);
    order[offs[e] + pos] = g;          // pair id p; token = p>>2
  }
}

// --------------------------------------------------- GEMM1: gate_up + act
__global__ __launch_bounds__(NTH) void moe_gateup(
    const float* __restrict__ X, const float* __restrict__ W1g,
    const float* __restrict__ B1g, const int* __restrict__ order,
    const Desc* __restrict__ desc, unsigned short* __restrict__ act) {
  const Desc d = desc[blockIdx.y];
  if (d.e < 0) return;
  const int e = d.e, row0 = d.row0, rowsv = d.rows;
  const int c0 = blockIdx.x * CHB;                       // first channel
  const float* __restrict__ W = W1g + (size_t)e * H_ * (2 * I_);

  const int tid  = threadIdx.x;
  const int lane = tid & 31;
  const int wm   = (tid >> 5) >> 1;                      // 0..7
  const int wn   = (tid >> 5) & 1;                       // 0..1

  __shared__ __align__(32) unsigned short Af[2][BM / 16][32][16];        // 32 KB
  __shared__ __align__(32) unsigned short Bf[2][(2 * CHB) / 16][32][16]; // 16 KB
  __shared__ int toks[BM];

  if (tid < BM) {
    int t = 0;
    if (tid < rowsv) t = order[row0 + tid] >> 2;
    toks[tid] = t;
  }
  __syncthreads();

  auto load_slab = [&](int kb, int buf) {
    const int k0 = kb * KT;
    // W1 slab: 32 k-rows x 128 cols (gate/up interleaved -> de-interleave into q)
#pragma unroll
    for (int ii = 0; ii < 2; ++ii) {
      const int f = tid + ii * NTH;                      // 0..1023 float4 id
      const int k = f >> 5;
      const int r = (f & 31) << 2;
      const float4 v = *(const float4*)(W + (size_t)(k0 + k) * (2 * I_) + 2 * c0 + r);
      const float vv[4] = {v.x, v.y, v.z, v.w};
      const int lbase = 16 * (k >> 4);
      const int i = k & 15;
#pragma unroll
      for (int j = 0; j < 4; ++j) {
        const int rr = r + j;
        const int chl = rr >> 1, part = rr & 1;
        const int q = ((chl >> 5) << 6) + (part << 5) + (chl & 31);
        Bf[buf][q >> 4][(q & 15) + lbase][i] = f2bf(vv[j]);
      }
    }
    // X slab: 256 gathered token rows x 32 k
#pragma unroll
    for (int ii = 0; ii < 4; ++ii) {
      const int f = tid + ii * NTH;                      // 0..2047 float4 id
      const int m = f >> 3;
      const int kq = (f & 7) << 2;
      const float4 v = *(const float4*)(X + (size_t)toks[m] * H_ + k0 + kq);
      const int l  = (m & 15) + 16 * (kq >> 4);
      const int i0 = kq & 15;
      ushort4 h;
      h.x = f2bf(v.x); h.y = f2bf(v.y); h.z = f2bf(v.z); h.w = f2bf(v.w);
      *(ushort4*)&Af[buf][m >> 4][l][i0] = h;
    }
  };

  const v8f zero = {0.f, 0.f, 0.f, 0.f, 0.f, 0.f, 0.f, 0.f};
  v8f acc[2][4];
#pragma unroll
  for (int t = 0; t < 2; ++t)
#pragma unroll
    for (int n = 0; n < 4; ++n) acc[t][n] = zero;

  const int NK = H_ / KT;                                 // 90
  load_slab(0, 0);
  for (int kb = 0; kb < NK; ++kb) {
    __syncthreads();
    if (kb + 1 < NK) load_slab(kb + 1, (kb + 1) & 1);
    const int buf = kb & 1;
    v16bf a0 = *(const v16bf*)&Af[buf][wm * 2 + 0][lane][0];
    v16bf a1 = *(const v16bf*)&Af[buf][wm * 2 + 1][lane][0];
#pragma unroll
    for (int n = 0; n < 4; ++n) {
      v16bf b = *(const v16bf*)&Bf[buf][wn * 4 + n][lane][0];
      acc[0][n] = __builtin_amdgcn_wmma_f32_16x16x32_bf16(false, a0, false, b, (short)0, acc[0][n], false, false);
      acc[1][n] = __builtin_amdgcn_wmma_f32_16x16x32_bf16(false, a1, false, b, (short)0, acc[1][n], false, false);
    }
  }

  // epilogue: gate/up land in identical lanes -> pure register activation
  const float* __restrict__ b1 = B1g + (size_t)e * (2 * I_);
  const int rsel = (lane >> 4) << 3;
#pragma unroll
  for (int t = 0; t < 2; ++t) {
    const int mt = (wm * 2 + t) * 16;
#pragma unroll
    for (int ct = 0; ct < 2; ++ct) {
      const int ch = c0 + wn * 32 + ct * 16 + (lane & 15);
      const float bg = b1[2 * ch], bu = b1[2 * ch + 1];
      const v8f g = acc[t][ct];
      const v8f u = acc[t][2 + ct];
#pragma unroll
      for (int r = 0; r < 8; ++r) {
        const int ml = mt + rsel + r;
        if (ml < rowsv) {
          float gv = fminf(g[r] + bg, LIMIT);
          float uv = fminf(fmaxf(u[r] + bu, -LIMIT), LIMIT);
          float s  = 1.0f / (1.0f + __expf(-ALPHA * gv));
          float val = (uv + 1.0f) * (gv * s);
          act[(size_t)(row0 + ml) * I_ + ch] = f2bf(val);
        }
      }
    }
  }
}

// --------------------------------------------------- GEMM2: down + scatter
__global__ __launch_bounds__(NTH) void moe_down(
    const unsigned short* __restrict__ act, const float* __restrict__ W2g,
    const float* __restrict__ B2g, const int* __restrict__ order,
    const float* __restrict__ rw, const Desc* __restrict__ desc,
    float* __restrict__ out) {
  const Desc d = desc[blockIdx.y];
  if (d.e < 0) return;
  const int e = d.e, row0 = d.row0, rowsv = d.rows;
  const int n0 = blockIdx.x * BN2;
  const float* __restrict__ W = W2g + (size_t)e * I_ * H_;

  const int tid  = threadIdx.x;
  const int lane = tid & 31;
  const int wm   = (tid >> 5) >> 1;                      // 0..7
  const int wn   = (tid >> 5) & 1;                       // 0..1

  __shared__ __align__(32) unsigned short Af[2][BM / 16][32][16];   // 32 KB
  __shared__ __align__(32) unsigned short Bf[2][BN2 / 16][32][16];  // 8 KB
  __shared__ int ords[BM];

  if (tid < BM) ords[tid] = (tid < rowsv) ? order[row0 + tid] : 0;
  __syncthreads();

  auto load_slab = [&](int kb, int buf) {
    const int k0 = kb * KT;
    // W2 slab: 32 x 64 fp32
    {
      const int f = tid;                                 // 0..511 float4 id
      const int k = f >> 4;
      const int r = (f & 15) << 2;
      const float4 v = *(const float4*)(W + (size_t)(k0 + k) * H_ + n0 + r);
      const float vv[4] = {v.x, v.y, v.z, v.w};
      const int lbase = 16 * (k >> 4);
      const int i = k & 15;
#pragma unroll
      for (int j = 0; j < 4; ++j) {
        const int q = r + j;
        Bf[buf][q >> 4][(q & 15) + lbase][i] = f2bf(vv[j]);
      }
    }
    // act slab: 256 rows x 32 bf16 (raw bf16 16B chunks) -> async copy to LDS
#pragma unroll
    for (int ii = 0; ii < 2; ++ii) {
      const int f = tid + ii * NTH;                      // 0..1023 chunk id
      const int m = f >> 2;
      const int c8 = (f & 3) << 3;
      size_t arow = (size_t)row0 + m;
      if (arow >= TKP) arow = TKP - 1;                   // clamp (masked later)
      const unsigned short* gp = act + arow * I_ + k0 + c8;
      const int l  = (m & 15) + 16 * (c8 >> 4);
      const int i0 = c8 & 15;
#if USE_ASYNC_LDS
      unsigned ldsaddr = (unsigned)(unsigned long long)&Af[buf][m >> 4][l][i0];
      asm volatile("global_load_async_to_lds_b128 %0, %1, off"
                   :: "v"(ldsaddr), "v"(gp) : "memory");
#else
      const uint4 v = *(const uint4*)gp;
      *(uint4*)&Af[buf][m >> 4][l][i0] = v;
#endif
    }
  };

  const v8f zero = {0.f, 0.f, 0.f, 0.f, 0.f, 0.f, 0.f, 0.f};
  v8f acc[2][2] = {{zero, zero}, {zero, zero}};

  const int NK = I_ / KT;                                 // 90
  load_slab(0, 0);
  for (int kb = 0; kb < NK; ++kb) {
#if USE_ASYNC_LDS
    asm volatile("s_wait_asynccnt 0x0" ::: "memory");     // drain async LDS writes
#endif
    __syncthreads();
    if (kb + 1 < NK) load_slab(kb + 1, (kb + 1) & 1);
    const int buf = kb & 1;
    v16bf a0 = *(const v16bf*)&Af[buf][wm * 2 + 0][lane][0];
    v16bf a1 = *(const v16bf*)&Af[buf][wm * 2 + 1][lane][0];
#pragma unroll
    for (int n = 0; n < 2; ++n) {
      v16bf b = *(const v16bf*)&Bf[buf][wn * 2 + n][lane][0];
      acc[0][n] = __builtin_amdgcn_wmma_f32_16x16x32_bf16(false, a0, false, b, (short)0, acc[0][n], false, false);
      acc[1][n] = __builtin_amdgcn_wmma_f32_16x16x32_bf16(false, a1, false, b, (short)0, acc[1][n], false, false);
    }
  }

  const float* __restrict__ b2 = B2g + (size_t)e * H_;
  const int rsel = (lane >> 4) << 3;
#pragma unroll
  for (int t = 0; t < 2; ++t) {
    const int mt = (wm * 2 + t) * 16;
#pragma unroll
    for (int n = 0; n < 2; ++n) {
      const int col = n0 + wn * 32 + n * 16 + (lane & 15);
      const float bb = b2[col];
      const v8f c = acc[t][n];
#pragma unroll
      for (int r = 0; r < 8; ++r) {
        const int ml = mt + rsel + r;
        if (ml < rowsv) {
          const int p = ords[ml];
          atomicAdd(out + (size_t)(p >> 2) * H_ + col, rw[p] * (c[r] + bb));
        }
      }
    }
  }
}

// ---------------------------------------------------------------- launcher
extern "C" void kernel_launch(void* const* d_in, const int* in_sizes, int n_in,
                              void* d_out, int out_size, void* d_ws, size_t ws_size,
                              hipStream_t stream) {
  (void)in_sizes; (void)n_in; (void)out_size; (void)ws_size;
  const float* X  = (const float*)d_in[0];
  const int*   RI = (const int*)d_in[1];
  const float* RW = (const float*)d_in[2];
  const float* W1 = (const float*)d_in[3];
  const float* B1 = (const float*)d_in[4];
  const float* W2 = (const float*)d_in[5];
  const float* B2 = (const float*)d_in[6];
  float* out = (float*)d_out;

  char* ws = (char*)d_ws;
  int*  cnt    = (int*)(ws + 0);            // 16
  int*  cursor = (int*)(ws + 64);           // 16
  int*  offs   = (int*)(ws + 128);          // 16
  Desc* desc   = (Desc*)(ws + 256);         // MAXD * 16 B
  int*  order  = (int*)(ws + 8192);         // 32768 * 4 B
  unsigned short* act = (unsigned short*)(ws + (1 << 20)); // 32768*2880*2 B ~ 189 MB

  k_zero_out<<<(T_ * H_ / 4) / 256, 256, 0, stream>>>((float4*)out);
  k_zero_misc<<<1, 64, 0, stream>>>(cnt);
  k_count<<<TKP / 256, 256, 0, stream>>>(RI, cnt);
  k_prefix<<<1, 1, 0, stream>>>(cnt, offs, desc);
  k_scatter<<<TKP / 256, 256, 0, stream>>>(RI, offs, cursor, order);

  dim3 g1(I_ / CHB, MAXD);   // 45 x 160
  moe_gateup<<<g1, NTH, 0, stream>>>(X, W1, B1, order, desc, act);
  dim3 g2(H_ / BN2, MAXD);   // 45 x 160
  moe_down<<<g2, NTH, 0, stream>>>(act, W2, B2, order, RW, desc, out);
}
